// MAEEGReconstruction_77515569758307
// MI455X (gfx1250) — matrix operand order repaired and checked by reading, not compiled, not run on hardware
//
#include <hip/hip_runtime.h>

// ---------------------------------------------------------------------------
// MAE EEG reconstruction for MI455X (gfx1250, wave32, WMMA).
// All matmuls (convs via im2col, QKV/O, QK^T, PV, FF1/FF2) run on
// v_wmma_f32_16x16x32_f16. f16-native operands are staged into LDS with
// GLOBAL_LOAD_ASYNC_TO_LDS_B128 (ASYNCcnt) when the toolchain exposes it;
// fp32 operands are converted to f16 during staging with packed cvt +
// ds_store_b128.
// ---------------------------------------------------------------------------

typedef __attribute__((ext_vector_type(16))) _Float16 v16h;
typedef __attribute__((ext_vector_type(8)))  _Float16 v8h;
typedef __attribute__((ext_vector_type(8)))  float    v8f;
typedef __attribute__((ext_vector_type(4)))  float    f4;
typedef __attribute__((ext_vector_type(4)))  int      i4;

union HU { v16h v; v8h h[2]; };

static constexpr int kB    = 16;
static constexpr int kCIN  = 64;
static constexpr int kT    = 1024;
static constexpr int kS    = 512;    // sequence after stride-2 conv
static constexpr int kD    = 512;
static constexpr int kH    = 8;
static constexpr int kHD   = 64;
static constexpr int kFF   = 2048;
static constexpr int kNL   = 8;
static constexpr int kM    = kB * kS;   // 8192 rows in the activation stream
static constexpr float kEPS = 1e-5f;

#if defined(__gfx1250__) && __has_builtin(__builtin_amdgcn_global_load_async_to_lds_b128)
#define HAS_ASYNC_LDS 1
typedef __attribute__((address_space(1))) i4 gi4;   // global int4 (HIP "__device__")
typedef __attribute__((address_space(3))) i4 li4;   // LDS int4
#else
#define HAS_ASYNC_LDS 0
#endif

__device__ __forceinline__ void wait_async_lds() {
#if HAS_ASYNC_LDS
#if __has_builtin(__builtin_amdgcn_s_wait_asynccnt)
    __builtin_amdgcn_s_wait_asynccnt(0);
#else
    asm volatile("s_wait_asynccnt 0" ::: "memory");
#endif
#endif
}

// --------------------------- WMMA GEMM ------------------------------------
// C[M,N] (+bias, relu) = alpha * A[M,K] * B[N,K]^T   (B stored row-major [N,K])
// Block tile 128x128, K-tile 32, 256 threads = 8 waves (2x4), double-buffered LDS.
#define TM 128
#define TN 128
#define TK 32
#define LDT 40   // LDS row stride in halves (padding vs 64-bank conflicts)

// Branch-free staging: out-of-range rows are clamped (their products only land in
// output elements the epilogue masks), so no zero-fill / exec-mask churn.
template<bool F16>
__device__ __forceinline__ void stage_tile(const void* __restrict__ src, long long off,
                                           int row0, int nrows, int ld, int kt,
                                           int r, int seg, _Float16* __restrict__ dst)
{
    const int k0 = kt * TK + seg * 16;
    _Float16* d = dst + r * LDT + seg * 16;
    int rr = row0 + r;
    rr = (rr < nrows) ? rr : (nrows - 1);
    if (F16) {
        const _Float16* s = (const _Float16*)src + off + (long long)rr * ld + k0;
#if HAS_ASYNC_LDS
        _Float16* sm = const_cast<_Float16*>(s);
        __builtin_amdgcn_global_load_async_to_lds_b128((gi4*)sm,       (li4*)d,       0, 0);
        __builtin_amdgcn_global_load_async_to_lds_b128((gi4*)(sm + 8), (li4*)(d + 8), 0, 0);
#else
        *(v8h*)d       = *(const v8h*)s;
        *(v8h*)(d + 8) = *(const v8h*)(s + 8);
#endif
    } else {
        const float* s = (const float*)src + off + (long long)rr * ld + k0;
        f4 q0 = *(const f4*)s;       f4 q1 = *(const f4*)(s + 4);
        f4 q2 = *(const f4*)(s + 8); f4 q3 = *(const f4*)(s + 12);
        v8h lo, hi;
        lo[0] = (_Float16)q0.x; lo[1] = (_Float16)q0.y; lo[2] = (_Float16)q0.z; lo[3] = (_Float16)q0.w;
        lo[4] = (_Float16)q1.x; lo[5] = (_Float16)q1.y; lo[6] = (_Float16)q1.z; lo[7] = (_Float16)q1.w;
        hi[0] = (_Float16)q2.x; hi[1] = (_Float16)q2.y; hi[2] = (_Float16)q2.z; hi[3] = (_Float16)q2.w;
        hi[4] = (_Float16)q3.x; hi[5] = (_Float16)q3.y; hi[6] = (_Float16)q3.z; hi[7] = (_Float16)q3.w;
        *(v8h*)d       = lo;
        *(v8h*)(d + 8) = hi;
    }
}

// CMODE: 0 = normal C store (OF16 selects f16/f32), 1 = V^T scatter (f16)
template<bool AF16, bool BF16, bool OF16, int CMODE, bool BIAS, bool RELU>
__global__ __launch_bounds__(256)
void gemm_wmma(const void* __restrict__ Ag, const void* __restrict__ Bg,
               void* __restrict__ Cg, const float* __restrict__ bias,
               int M, int N, int K, int lda, int ldb, int ldc, float alpha,
               int shA, long long sA1, long long sA2,
               int shB, long long sB1, long long sB2,
               int shC, long long sC1, long long sC2)
{
    __shared__ __align__(16) _Float16 lA[2][TM * LDT];
    __shared__ __align__(16) _Float16 lB[2][TN * LDT];

    const int tid = threadIdx.x;
    const int z   = blockIdx.z;
    const long long offA = ((long long)(z >> shA)) * sA1 + (long long)(z & ((1 << shA) - 1)) * sA2;
    const long long offB = ((long long)(z >> shB)) * sB1 + (long long)(z & ((1 << shB) - 1)) * sB2;
    const long long offC = ((long long)(z >> shC)) * sC1 + (long long)(z & ((1 << shC) - 1)) * sC2;

    const int m0 = blockIdx.y * TM;
    const int n0 = blockIdx.x * TN;

    // staging role: 128 rows x 2 segments of 16 elements
    const int sr  = tid >> 1;
    const int seg = tid & 1;
    // wave role
    const int wid  = tid >> 5;
    const int lane = tid & 31;
    const int wm   = wid >> 2;   // 0..1 -> 64 rows each
    const int wn   = wid & 3;    // 0..3 -> 32 cols each
    const int mrow = lane & 15;
    const int hs   = lane >> 4;

    v8f acc[4][2];
#pragma unroll
    for (int i = 0; i < 4; ++i)
#pragma unroll
        for (int j = 0; j < 2; ++j)
#pragma unroll
            for (int r = 0; r < 8; ++r) acc[i][j][r] = 0.f;

    const int nk = K / TK;

    stage_tile<AF16>(Ag, offA, m0, M, lda, 0, sr, seg, &lA[0][0]);
    stage_tile<BF16>(Bg, offB, n0, N, ldb, 0, sr, seg, &lB[0][0]);
    if (AF16 || BF16) wait_async_lds();
    __syncthreads();

    for (int kt = 0; kt < nk; ++kt) {
        const bool more = (kt + 1 < nk);
        if (more) {
            stage_tile<AF16>(Ag, offA, m0, M, lda, kt + 1, sr, seg, &lA[(kt + 1) & 1][0]);
            stage_tile<BF16>(Bg, offB, n0, N, ldb, kt + 1, sr, seg, &lB[(kt + 1) & 1][0]);
        }
        const _Float16* bA = &lA[kt & 1][0];
        const _Float16* bB = &lB[kt & 1][0];

        v16h af[4], bf[2];
#pragma unroll
        for (int i = 0; i < 4; ++i) {
            const _Float16* p = bA + (wm * 64 + i * 16 + mrow) * LDT;
            HU u;
            u.h[0] = *(const v8h*)(p + hs * 8);
            u.h[1] = *(const v8h*)(p + hs * 8 + 16);
            af[i] = u.v;
        }
#pragma unroll
        for (int j = 0; j < 2; ++j) {
            const _Float16* p = bB + (wn * 32 + j * 16 + mrow) * LDT;
            HU u;
            u.h[0] = *(const v8h*)(p + hs * 8);
            u.h[1] = *(const v8h*)(p + hs * 8 + 16);
            bf[j] = u.v;
        }
#pragma unroll
        for (int i = 0; i < 4; ++i)
#pragma unroll
            for (int j = 0; j < 2; ++j)
                acc[i][j] = __builtin_amdgcn_wmma_f32_16x16x32_f16(
                    false, af[i], false, bf[j], (short)0, acc[i][j], false, false);

        if (more && (AF16 || BF16)) wait_async_lds();
        __syncthreads();
    }

    // epilogue; C fragment layout: VGPR r -> row r (lanes 0-15) / row 8+r (lanes 16-31), col = lane&15
    const bool full = (m0 + TM <= M) && (n0 + TN <= N);
    if (CMODE == 0 && full) {
#pragma unroll
        for (int i = 0; i < 4; ++i) {
            const int rbase = m0 + wm * 64 + i * 16 + ((lane < 16) ? 0 : 8);
#pragma unroll
            for (int j = 0; j < 2; ++j) {
                const int col = n0 + wn * 32 + j * 16 + mrow;
                const float bv = BIAS ? bias[col] : 0.f;
#pragma unroll
                for (int r = 0; r < 8; ++r) {
                    float v = acc[i][j][r] * alpha;
                    if (BIAS) v += bv;
                    if (RELU) v = fmaxf(v, 0.f);
                    const long long ci = offC + (long long)(rbase + r) * ldc + col;
                    if (OF16) ((_Float16*)Cg)[ci] = (_Float16)v;
                    else      ((float*)Cg)[ci]    = v;
                }
            }
        }
    } else {
#pragma unroll
        for (int i = 0; i < 4; ++i) {
            const int rbase = m0 + wm * 64 + i * 16 + ((lane < 16) ? 0 : 8);
#pragma unroll
            for (int j = 0; j < 2; ++j) {
                const int col = n0 + wn * 32 + j * 16 + mrow;
                const float bv = (BIAS && col < N) ? bias[col] : 0.f;
#pragma unroll
                for (int r = 0; r < 8; ++r) {
                    const int row = rbase + r;
                    if (row < M && col < N) {
                        float v = acc[i][j][r] * alpha;
                        if (BIAS) v += bv;
                        if (RELU) v = fmaxf(v, 0.f);
                        if (CMODE == 0) {
                            const long long ci = offC + (long long)row * ldc + col;
                            if (OF16) ((_Float16*)Cg)[ci] = (_Float16)v;
                            else      ((float*)Cg)[ci]    = v;
                        } else {
                            // V^T scatter: row=(b*512+s), col=h*64+u -> Vt[((b*8+h)*64+u)*512 + s]
                            const int b  = row >> 9, s = row & 511;
                            const int hh = col >> 6, u = col & 63;
                            ((_Float16*)Cg)[(((long long)(b * kH + hh) * kHD + u) << 9) + s] = (_Float16)v;
                        }
                    }
                }
            }
        }
    }
}

// --------------------------- small kernels ---------------------------------

__device__ __forceinline__ float gelu_exact(float x) {
    return 0.5f * x * (1.f + erff(x * 0.70710678118654752f));
}

// im2col for conv0: x[16,64,1024] -> ic0[8192, 960] f16, col = c*15 + k, src t = 2*tout-7+k
__global__ void im2col0_kernel(const float* __restrict__ x, _Float16* __restrict__ ic0)
{
    int idx = blockIdx.x * blockDim.x + threadIdx.x;   // 8192*64
    int row = idx >> 6, c = idx & 63;
    int b = row >> 9, t = row & 511;
    const float* xp = x + ((long long)(b * kCIN + c)) * kT;
    _Float16* o = ic0 + (long long)row * 960 + c * 15;
    int s0 = 2 * t - 7;
#pragma unroll
    for (int k = 0; k < 15; ++k) {
        int s = s0 + k;
        o[k] = (s >= 0 && s < kT) ? (_Float16)xp[s] : (_Float16)0.f;
    }
}

// im2col for conv1: h0[16,512,512] (b,t,c) -> ic1[8192,1536] f16, col = k*512 + c, src t' = t-1+k
__global__ void im2col1_kernel(const float* __restrict__ h0, _Float16* __restrict__ ic1)
{
    long long idx = (long long)blockIdx.x * blockDim.x + threadIdx.x;   // 8192*1536
    int row = (int)(idx / 1536), j = (int)(idx % 1536);
    int k = j >> 9, c = j & 511;
    int b = row >> 9, t = row & 511;
    int s = t - 1 + k;
    ic1[idx] = (s >= 0 && s < kS) ? (_Float16)h0[(((long long)(b << 9) + s) << 9) + c]
                                  : (_Float16)0.f;
}

// permute conv1 weights [co][ci][3] -> f16 [co][k*512+ci]
__global__ void permW1_kernel(const float* __restrict__ w, _Float16* __restrict__ wp)
{
    int idx = blockIdx.x * blockDim.x + threadIdx.x;   // 512*1536
    int co = idx / 1536, j = idx % 1536;
    int k = j >> 9, c = j & 511;
    wp[idx] = (_Float16)w[((long long)(co * kD + c)) * 3 + k];
}

// GroupNorm(groups=256 on layout [b][t][c]) + exact GELU, in place. One block per (b,group).
__global__ __launch_bounds__(256) void gn_gelu_kernel(float* __restrict__ h,
                                                      const float* __restrict__ g,
                                                      const float* __restrict__ bta)
{
    const int bg = blockIdx.x;
    const int b = bg >> 8, grp = bg & 255;
    const int tid = threadIdx.x;
    const long long base = ((long long)b * kS) * kD;

    float v[4]; long long ix[4]; int cc[4];
    float s = 0.f, q = 0.f;
#pragma unroll
    for (int i = 0; i < 4; ++i) {
        int e = tid + 256 * i;          // 1024 elems: 2 channels x 512 positions
        int t = e >> 1, c = grp * 2 + (e & 1);
        long long idx = base + (long long)t * kD + c;
        float vv = h[idx];
        v[i] = vv; ix[i] = idx; cc[i] = c;
        s += vv; q += vv * vv;
    }
#pragma unroll
    for (int o = 16; o; o >>= 1) { s += __shfl_xor(s, o, 32); q += __shfl_xor(q, o, 32); }
    __shared__ float rs[8], rq[8];
    const int wid = tid >> 5, lane = tid & 31;
    if (lane == 0) { rs[wid] = s; rq[wid] = q; }
    __syncthreads();
    float S = 0.f, Q = 0.f;
#pragma unroll
    for (int w = 0; w < 8; ++w) { S += rs[w]; Q += rq[w]; }
    const float mu   = S * (1.f / 1024.f);
    const float var  = Q * (1.f / 1024.f) - mu * mu;
    const float rstd = rsqrtf(var + kEPS);
#pragma unroll
    for (int i = 0; i < 4; ++i) {
        float y = (v[i] - mu) * rstd * g[cc[i]] + bta[cc[i]];
        h[ix[i]] = gelu_exact(y);
    }
}

// out = LayerNorm(x + r) * g + b over rows of 512. 256 threads = 8 waves, 1 row/wave.
__global__ __launch_bounds__(256) void ln_res_kernel(const float* __restrict__ x,
                                                     const float* __restrict__ r,
                                                     const float* __restrict__ g,
                                                     const float* __restrict__ b,
                                                     float* __restrict__ out, int rows)
{
    const int row = blockIdx.x * 8 + (threadIdx.x >> 5);
    const int lane = threadIdx.x & 31;
    if (row >= rows) return;
    const float* px = x + (long long)row * kD;
    const float* pr = r + (long long)row * kD;
    float v[16]; float s = 0.f;
#pragma unroll
    for (int i = 0; i < 16; ++i) { v[i] = px[lane + i * 32] + pr[lane + i * 32]; s += v[i]; }
#pragma unroll
    for (int o = 16; o; o >>= 1) s += __shfl_xor(s, o, 32);
    const float mu = s * (1.f / 512.f);
    float q = 0.f;
#pragma unroll
    for (int i = 0; i < 16; ++i) { float d = v[i] - mu; q += d * d; }
#pragma unroll
    for (int o = 16; o; o >>= 1) q += __shfl_xor(q, o, 32);
    const float rstd = rsqrtf(q * (1.f / 512.f) + kEPS);
    float* po = out + (long long)row * kD;
#pragma unroll
    for (int i = 0; i < 16; ++i) {
        int c = lane + i * 32;
        po[c] = (v[i] - mu) * rstd * g[c] + b[c];
    }
}

// in-place row softmax over f16 rows of length 512 (scores). 1 row/wave.
__global__ __launch_bounds__(256) void softmax_f16_kernel(_Float16* __restrict__ p, int rows)
{
    const int row = blockIdx.x * 8 + (threadIdx.x >> 5);
    const int lane = threadIdx.x & 31;
    if (row >= rows) return;
    _Float16* pr = p + (long long)row * kS;
    float v[16]; float m = -3.4e38f;
#pragma unroll
    for (int i = 0; i < 16; ++i) { v[i] = (float)pr[lane + i * 32]; m = fmaxf(m, v[i]); }
#pragma unroll
    for (int o = 16; o; o >>= 1) m = fmaxf(m, __shfl_xor(m, o, 32));
    float s = 0.f;
#pragma unroll
    for (int i = 0; i < 16; ++i) { v[i] = __expf(v[i] - m); s += v[i]; }
#pragma unroll
    for (int o = 16; o; o >>= 1) s += __shfl_xor(s, o, 32);
    const float inv = 1.f / s;
#pragma unroll
    for (int i = 0; i < 16; ++i) pr[lane + i * 32] = (_Float16)(v[i] * inv);
}

// ConvTranspose1d(512->64, k=3, s=2, p=1, out_pad=1) from h[b][s][c] -> out[16,64,1024]
__global__ __launch_bounds__(256) void deconv_kernel(const float* __restrict__ h,
                                                     const float* __restrict__ Wd,
                                                     const float* __restrict__ bd,
                                                     float* __restrict__ out)
{
    int idx = blockIdx.x * blockDim.x + threadIdx.x;   // 16*64*1024
    int t  = idx & 1023;
    int co = (idx >> 10) & 63;
    int b  = idx >> 16;
    float acc = bd[co];
#pragma unroll
    for (int k = 0; k < 3; ++k) {
        int tp = t + 1 - k;
        if (tp >= 0 && (tp & 1) == 0) {
            int s = tp >> 1;
            if (s < kS) {
                const float* hr = h + ((long long)(b * kS + s) << 9);
                const float* wp = Wd + co * 3 + k;   // stride 192 over ci
                float a2 = 0.f;
                for (int ci = 0; ci < kD; ++ci) a2 += hr[ci] * wp[ci * 192];
                acc += a2;
            }
        }
    }
    out[idx] = acc;
}

// --------------------------- host launcher ---------------------------------

extern "C" void kernel_launch(void* const* d_in, const int* in_sizes, int n_in,
                              void* d_out, int out_size, void* d_ws, size_t ws_size,
                              hipStream_t stream)
{
    const float* x      = (const float*)d_in[0];
    const float* convW0 = (const float*)d_in[1];
    const float* convb0 = (const float*)d_in[2];
    const float* gn0_g  = (const float*)d_in[3];
    const float* gn0_b  = (const float*)d_in[4];
    const float* convW1 = (const float*)d_in[5];
    const float* convb1 = (const float*)d_in[6];
    const float* gn1_g  = (const float*)d_in[7];
    const float* gn1_b  = (const float*)d_in[8];
    const float* Wq     = (const float*)d_in[9];
    const float* Wk     = (const float*)d_in[10];
    const float* Wv     = (const float*)d_in[11];
    const float* Wo     = (const float*)d_in[12];
    const float* W1     = (const float*)d_in[13];
    const float* b1     = (const float*)d_in[14];
    const float* W2     = (const float*)d_in[15];
    const float* b2     = (const float*)d_in[16];
    const float* ln1_g  = (const float*)d_in[17];
    const float* ln1_b  = (const float*)d_in[18];
    const float* ln2_g  = (const float*)d_in[19];
    const float* ln2_b  = (const float*)d_in[20];
    const float* Wd     = (const float*)d_in[21];
    const float* bd     = (const float*)d_in[22];
    float* out = (float*)d_out;

    // workspace bump allocator (256B aligned)
    char* wsp = (char*)d_ws;
    auto alloc = [&](size_t bytes) -> void* {
        void* p = (void*)wsp;
        wsp += (bytes + 255) & ~(size_t)255;
        return p;
    };
    _Float16* ic0   = (_Float16*)alloc((size_t)kM * 960 * 2);
    float*    h0    = (float*)   alloc((size_t)kM * kD * 4);
    _Float16* W1p   = (_Float16*)alloc((size_t)kD * 1536 * 2);
    _Float16* ic1   = (_Float16*)alloc((size_t)kM * 1536 * 2);
    float*    hbuf  = (float*)   alloc((size_t)kM * kD * 4);
    _Float16* Qf    = (_Float16*)alloc((size_t)kM * kD * 2);
    _Float16* Kf    = (_Float16*)alloc((size_t)kM * kD * 2);
    _Float16* Vt    = (_Float16*)alloc((size_t)kM * kD * 2);
    _Float16* scor  = (_Float16*)alloc((size_t)kB * kH * kS * kS * 2);
    _Float16* attnC = (_Float16*)alloc((size_t)kM * kD * 2);
    float*    abuf  = (float*)   alloc((size_t)kM * kD * 4);
    float*    h1    = (float*)   alloc((size_t)kM * kD * 4);
    _Float16* ffmid = (_Float16*)alloc((size_t)kM * kFF * 2);
    float*    fbuf  = (float*)   alloc((size_t)kM * kD * 4);
    (void)ws_size; (void)n_in; (void)in_sizes; (void)out_size;

    const long long SD = (long long)kS * kD;          // 262144
    const dim3 blk(256);

    // ---- conv encoder ----
    im2col0_kernel<<<(kM * kCIN) / 256, blk, 0, stream>>>(x, ic0);
    gemm_wmma<true, false, false, 0, true, false><<<dim3(4, 64, 1), blk, 0, stream>>>(
        ic0, convW0, h0, convb0, kM, kD, 960, 960, 960, kD, 1.f,
        0, 0, 0, 0, 0, 0, 0, 0, 0);
    gn_gelu_kernel<<<kB * 256, blk, 0, stream>>>(h0, gn0_g, gn0_b);

    permW1_kernel<<<(kD * 1536) / 256, blk, 0, stream>>>(convW1, W1p);
    im2col1_kernel<<<(kM * 1536) / 256, blk, 0, stream>>>(h0, ic1);
    gemm_wmma<true, true, false, 0, true, false><<<dim3(4, 64, 1), blk, 0, stream>>>(
        ic1, W1p, hbuf, convb1, kM, kD, 1536, 1536, 1536, kD, 1.f,
        0, 0, 0, 0, 0, 0, 0, 0, 0);
    gn_gelu_kernel<<<kB * 256, blk, 0, stream>>>(hbuf, gn1_g, gn1_b);

    // ---- transformer ----
    for (int i = 0; i < kNL; ++i) {
        const float* Wq_i = Wq + (long long)i * kD * kD;
        const float* Wk_i = Wk + (long long)i * kD * kD;
        const float* Wv_i = Wv + (long long)i * kD * kD;
        const float* Wo_i = Wo + (long long)i * kD * kD;
        const float* W1_i = W1 + (long long)i * kFF * kD;
        const float* b1_i = b1 + (long long)i * kFF;
        const float* W2_i = W2 + (long long)i * kD * kFF;
        const float* b2_i = b2 + (long long)i * kD;

        // Q, K (f16), V (f16, transposed per head)
        gemm_wmma<false, false, true, 0, false, false><<<dim3(4, 64, 1), blk, 0, stream>>>(
            hbuf, Wq_i, Qf, nullptr, kM, kD, kD, kD, kD, kD, 1.f,
            0, 0, 0, 0, 0, 0, 0, 0, 0);
        gemm_wmma<false, false, true, 0, false, false><<<dim3(4, 64, 1), blk, 0, stream>>>(
            hbuf, Wk_i, Kf, nullptr, kM, kD, kD, kD, kD, kD, 1.f,
            0, 0, 0, 0, 0, 0, 0, 0, 0);
        gemm_wmma<false, false, true, 1, false, false><<<dim3(4, 64, 1), blk, 0, stream>>>(
            hbuf, Wv_i, Vt, nullptr, kM, kD, kD, kD, kD, kD, 1.f,
            0, 0, 0, 0, 0, 0, 0, 0, 0);

        // scores[z] = (Q_z K_z^T) / 8, batched over z=(b*8+h)
        gemm_wmma<true, true, true, 0, false, false><<<dim3(4, 4, kB * kH), blk, 0, stream>>>(
            Qf, Kf, scor, nullptr, kS, kS, kHD, kD, kD, kS, 0.125f,
            3, SD, kHD, 3, SD, kHD, 0, (long long)kS * kS, 0);

        softmax_f16_kernel<<<(kB * kH * kS) / 8, blk, 0, stream>>>(scor, kB * kH * kS);

        // attnC[b,s,h*64+u] = P_z @ V_z   (B operand = Vt, contiguous in K)
        gemm_wmma<true, true, true, 0, false, false><<<dim3(1, 4, kB * kH), blk, 0, stream>>>(
            scor, Vt, attnC, nullptr, kS, kHD, kS, kS, kS, kD, 1.f,
            0, (long long)kS * kS, 0, 0, (long long)kHD * kS, 0, 3, SD, kHD);

        // O projection -> abuf (f32)
        gemm_wmma<true, false, false, 0, false, false><<<dim3(4, 64, 1), blk, 0, stream>>>(
            attnC, Wo_i, abuf, nullptr, kM, kD, kD, kD, kD, kD, 1.f,
            0, 0, 0, 0, 0, 0, 0, 0, 0);

        // h1 = LN(attn_out + h)
        ln_res_kernel<<<kM / 8, blk, 0, stream>>>(abuf, hbuf, ln1_g + i * kD, ln1_b + i * kD, h1, kM);

        // FF1: relu(h1 @ W1^T + b1) -> f16
        gemm_wmma<false, false, true, 0, true, true><<<dim3(16, 64, 1), blk, 0, stream>>>(
            h1, W1_i, ffmid, b1_i, kM, kFF, kD, kD, kD, kFF, 1.f,
            0, 0, 0, 0, 0, 0, 0, 0, 0);
        // FF2: ffmid @ W2^T + b2 -> f32
        gemm_wmma<true, false, false, 0, true, false><<<dim3(4, 64, 1), blk, 0, stream>>>(
            ffmid, W2_i, fbuf, b2_i, kM, kD, kFF, kFF, kFF, kD, 1.f,
            0, 0, 0, 0, 0, 0, 0, 0, 0);

        // h = LN(f + h1)
        ln_res_kernel<<<kM / 8, blk, 0, stream>>>(fbuf, h1, ln2_g + i * kD, ln2_b + i * kD, hbuf, kM);
    }

    // ---- deconv decoder ----
    deconv_kernel<<<(kB * kCIN * kT) / 256, blk, 0, stream>>>(hbuf, Wd, bd, out);
}
